// LorentzMultiHeadAttention_4337916969590
// MI455X (gfx1250) — compile-verified, matrix operand order checked by reading, not run on hardware
//
#include <hip/hip_runtime.h>

// ---------------------------------------------------------------------------
// CDNA5 (gfx1250) Lorentz multi-head attention, bf16 WMMA path +
// async global->LDS double-buffered K staging (ASYNCcnt) + NT probs stores.
// ---------------------------------------------------------------------------

typedef __attribute__((ext_vector_type(16))) __bf16 v16bf;
typedef __attribute__((ext_vector_type(8)))  float  v8f;
typedef __attribute__((ext_vector_type(4)))  float  f4;

__device__ __forceinline__ float  ldf(const float*  p) { return *p; }
__device__ __forceinline__ float  ldf(const __bf16* p) { return (float)(*p); }
__device__ __forceinline__ __bf16 f2bf(float f)        { return (__bf16)f; }
__device__ __forceinline__ void   stf(float*  p, float v) { *p = v; }
__device__ __forceinline__ void   stf(__bf16* p, float v) { *p = (__bf16)v; }

// ---------------------------------------------------------------------------
// Y[M,N] = X[M,K] @ W[N,K]^T (torch Linear). One 16x16 tile per wave,
// 4 waves/block covering 64 output columns. bf16 WMMA, f32 accumulate.
// Lane layout per ISA 7.12.2 (16-bit operands).
// ---------------------------------------------------------------------------
template <typename Tin, typename Tout>
__global__ void gemm_xwT_wmma(const Tin* __restrict__ X, const float* __restrict__ W,
                              Tout* __restrict__ Y, int M, int N, int K)
{
    const int lane  = threadIdx.x & 31;
    const int wave  = threadIdx.x >> 5;
    const int n0    = blockIdx.x * 64 + wave * 16;
    const int m0    = blockIdx.y * 16;
    const int khalf = lane >> 4;
    const int mr    = lane & 15;
    const int base  = khalf * 8;

    const Tin*   xrow = X + (size_t)(m0 + mr) * K;
    const float* wrow = W + (size_t)(n0 + mr) * K;

    v8f c = {};
    for (int kt = 0; kt < K; kt += 32) {
        v16bf a, b;
#pragma unroll
        for (int e = 0; e < 8; ++e) {
            a[e]     = f2bf(ldf(xrow + kt + base + e));
            a[e + 8] = f2bf(ldf(xrow + kt + 16 + base + e));
            b[e]     = f2bf(wrow[kt + base + e]);
            b[e + 8] = f2bf(wrow[kt + 16 + base + e]);
        }
        c = __builtin_amdgcn_wmma_f32_16x16x32_bf16(false, a, false, b,
                                                    (short)0, c, false, false);
    }
#pragma unroll
    for (int v = 0; v < 8; ++v)
        stf(Y + (size_t)(m0 + v + 8 * khalf) * N + (n0 + mr), c[v]);
}

// ---------------------------------------------------------------------------
// Q_mod = Q * (1 - 2*alpha*sf*mask), sf = |Q| / max(|Q*mask|, eps) per
// (b,l,h) head-row. One wave32 per row; 2 dims per lane; shuffle reduce.
// ---------------------------------------------------------------------------
__global__ void qmod_kernel(const __bf16* __restrict__ Q, const int* __restrict__ mask,
                            __bf16* __restrict__ Qmod, int BLH, int H, int dh, int D)
{
    const int gw   = (int)((blockIdx.x * blockDim.x + threadIdx.x) >> 5);
    const int lane = threadIdx.x & 31;
    if (gw >= BLH) return;
    const int h  = gw % H;
    const int bl = gw / H;

    const __bf16* q = Q    + (size_t)bl * D + h * dh;
    __bf16*       o = Qmod + (size_t)bl * D + h * dh;

    const int d0 = 2 * lane, d1 = 2 * lane + 1;
    const float q0 = (float)q[d0], q1 = (float)q[d1];
    const int   m0 = mask[h * dh + d0] != 0;
    const int   m1 = mask[h * dh + d1] != 0;

    float qs = q0 * q0 + q1 * q1;
    float qt = (m0 ? q0 * q0 : 0.f) + (m1 ? q1 * q1 : 0.f);
#pragma unroll
    for (int off = 16; off >= 1; off >>= 1) {
        qs += __shfl_xor(qs, off, 32);
        qt += __shfl_xor(qt, off, 32);
    }
    const float qn  = sqrtf(qs);
    const float qtn = sqrtf(qt);
    const float sf  = (qtn > 1e-6f) ? (qn / fmaxf(qtn, 1e-8f)) : 0.f;
    const float g   = 0.5f * sf;   // 2 * ALPHA(=0.25) * sf

    o[d0] = f2bf(q0 * (1.f - (m0 ? g : 0.f)));
    o[d1] = f2bf(q1 * (1.f - (m1 ? g : 0.f)));
}

// ---------------------------------------------------------------------------
// Vt[b,h,d,l] = V[b,l,h*dh+d]  (bf16) so P@V B-operand is contiguous in k.
// ---------------------------------------------------------------------------
__global__ void v_transpose(const __bf16* __restrict__ V, __bf16* __restrict__ Vt,
                            int B, int H, int L, int dh, int D)
{
    const int idx = blockIdx.x * blockDim.x + threadIdx.x;
    if (idx >= B * H * dh * L) return;
    const int l = idx % L;
    int t = idx / L;
    const int d = t % dh; t /= dh;
    const int h = t % H;
    const int b = t / H;
    Vt[idx] = V[(size_t)(b * L + l) * D + h * dh + d];
}

// ---------------------------------------------------------------------------
// Async DMA of one 16x64 bf16 K-tile (2 KB) into LDS.
// 4 issues x 32 lanes x 16 B: issue j moves rows j*4 .. j*4+3.
// lds_off / global addr are per-lane; tracked on ASYNCcnt.
// ---------------------------------------------------------------------------
__device__ __forceinline__ void async_copy_ktile(unsigned lds_off,
                                                 const __bf16* __restrict__ gbase,
                                                 int lane, size_t row_stride_bytes)
{
    const int rlane = lane >> 3;           // row within 4-row group
    const int coff  = (lane & 7) * 16;     // byte offset within 128 B row
#pragma unroll
    for (int j = 0; j < 4; ++j) {
        const int row = j * 4 + rlane;
        const char* g = (const char*)gbase + (size_t)row * row_stride_bytes + coff;
        unsigned    l = lds_off + (unsigned)(row * 128 + coff);
        asm volatile("global_load_async_to_lds_b128 %0, %1, off"
                     :: "v"(l), "v"(g) : "memory");
    }
}

// ---------------------------------------------------------------------------
// Attention core: one block per (b, h, 16-query tile). 8 waves, 256 threads.
// LDS: 16 x L f32 score strip (128 KB) + 8 waves x 2 x 2 KB K double-buffers
// (32 KB) = 160 KB of the CDNA5 320 KB/WGP LDS.
// Phase 1: scores = Qmod K^T / 8 via WMMA; Q tile hoisted to registers,
//          K tiles streamed by async global->LDS DMA, read back via DS.
// Phase 2: per-row softmax (16-lane groups). Phase 2b: NT probs store.
// Phase 3: O = P @ V via WMMA (waves 0..3 each own 16 head dims).
// ---------------------------------------------------------------------------
__global__ void lorentz_attn(const __bf16* __restrict__ Qm, const __bf16* __restrict__ Km,
                             const __bf16* __restrict__ Vt, float* __restrict__ probs,
                             __bf16* __restrict__ O, int B, int H, int L, int dh, int D)
{
    extern __shared__ char smem_raw[];
    float* s = (float*)smem_raw;                   // 16 * L floats
    const int nqt = L / 16;
    const int qt  = blockIdx.x % nqt;
    const int h   = (blockIdx.x / nqt) % H;
    const int b   = blockIdx.x / nqt / H;
    const int q0  = qt * 16;

    const int tid   = threadIdx.x;
    const int lane  = tid & 31;
    const int wave  = tid >> 5;
    const int khalf = lane >> 4;
    const int mr    = lane & 15;
    const int base  = khalf * 8;
    const float rscale = 0.125f;                   // 1/sqrt(dh), dh = 64

    // ---- Phase 1: scores into LDS, K tiles via async global->LDS DMA ----
    {
        // Hoist the loop-invariant 16x64 Q tile into registers (two WMMA
        // A-operands, kc = 0 and kc = 32) so the asm memory clobbers can't
        // force per-iteration reloads.
        const __bf16* qrow = Qm + (size_t)(b * L + q0 + mr) * D + h * dh;
        v16bf aq0, aq1;
#pragma unroll
        for (int e = 0; e < 8; ++e) {
            aq0[e]     = qrow[base + e];
            aq0[e + 8] = qrow[16 + base + e];
            aq1[e]     = qrow[32 + base + e];
            aq1[e + 8] = qrow[48 + base + e];
        }

        const unsigned kbuf0   = (unsigned)(16 * L * 4 + wave * 4096);
        const size_t   kstride = (size_t)D * sizeof(__bf16);
        const __bf16*  kg0     = Km + (size_t)(b * L) * D + h * dh;

        // prologue: stage first tile for this wave
        async_copy_ktile(kbuf0, kg0 + (size_t)(wave * 16) * D, lane, kstride);

        unsigned buf = 0;
        for (int kt = wave; kt < nqt; kt += 8) {
            const int k0  = kt * 16;
            const int ktn = kt + 8;
            if (ktn < nqt) {
                async_copy_ktile(kbuf0 + ((buf ^ 1u) << 11),
                                 kg0 + (size_t)(ktn * 16) * D, lane, kstride);
                asm volatile("s_wait_asynccnt 4" ::: "memory");  // current tile landed
            } else {
                asm volatile("s_wait_asynccnt 0" ::: "memory");
            }

            // B operand rows read straight from the staged LDS tile.
            // Single GEP chain rooted at the __shared__ symbol (offset
            // select, not pointer select) so it lowers to ds_load.
            const unsigned krow_off = kbuf0 + (buf << 11) + (unsigned)(mr * 128);
            v16bf bb0, bb1;
#pragma unroll
            for (int e = 0; e < 8; ++e) {
                bb0[e]     = *(const __bf16*)(smem_raw + krow_off + 2u * (base + e));
                bb0[e + 8] = *(const __bf16*)(smem_raw + krow_off + 2u * (16 + base + e));
                bb1[e]     = *(const __bf16*)(smem_raw + krow_off + 2u * (32 + base + e));
                bb1[e + 8] = *(const __bf16*)(smem_raw + krow_off + 2u * (48 + base + e));
            }

            v8f c = {};
            c = __builtin_amdgcn_wmma_f32_16x16x32_bf16(false, aq0, false, bb0,
                                                        (short)0, c, false, false);
            c = __builtin_amdgcn_wmma_f32_16x16x32_bf16(false, aq1, false, bb1,
                                                        (short)0, c, false, false);
#pragma unroll
            for (int v = 0; v < 8; ++v)
                s[(v + 8 * khalf) * L + k0 + mr] = c[v] * rscale;
            buf ^= 1u;
        }
    }
    __syncthreads();

    // ---- Phase 2: softmax, one row per 16-thread group ------------------
    {
        const int row = tid >> 4;
        const int idx = tid & 15;
        float* sr = s + row * L;
        float mx = -3.402823e38f;
        for (int j = idx; j < L; j += 16) mx = fmaxf(mx, sr[j]);
#pragma unroll
        for (int o = 8; o >= 1; o >>= 1) mx = fmaxf(mx, __shfl_xor(mx, o, 16));
        float sum = 0.f;
        for (int j = idx; j < L; j += 16) {
            float e = __expf(sr[j] - mx);
            sr[j] = e;
            sum += e;
        }
#pragma unroll
        for (int o = 8; o >= 1; o >>= 1) sum += __shfl_xor(sum, o, 16);
        const float inv = 1.0f / sum;
        for (int j = idx; j < L; j += 16) sr[j] *= inv;
    }
    __syncthreads();

    // ---- Phase 2b: non-temporal, vectorized probs store -----------------
    {
        f4*       pb4 = (f4*)(probs + ((size_t)(b * H + h) * L + q0) * L);
        const f4* s4  = (const f4*)s;
        for (int i = tid; i < 16 * L / 4; i += 256)
            __builtin_nontemporal_store(s4[i], pb4 + i);
    }

    // ---- Phase 3: O = P @ V, waves 0..3 each own 16 head dims -----------
    if (wave < 4) {
        const int n0 = wave * 16;
        const __bf16* vrow = Vt + (size_t)((b * H + h) * dh + n0 + mr) * L;
        v8f c = {};
        for (int kc = 0; kc < L; kc += 32) {
            v16bf a, bb;
#pragma unroll
            for (int e = 0; e < 8; ++e) {
                a[e]      = f2bf(s[mr * L + kc + base + e]);
                a[e + 8]  = f2bf(s[mr * L + kc + 16 + base + e]);
                bb[e]     = vrow[kc + base + e];
                bb[e + 8] = vrow[kc + 16 + base + e];
            }
            c = __builtin_amdgcn_wmma_f32_16x16x32_bf16(false, a, false, bb,
                                                        (short)0, c, false, false);
        }
#pragma unroll
        for (int v = 0; v < 8; ++v)
            O[(size_t)(b * L + q0 + v + 8 * khalf) * D + h * dh + n0 + mr] = f2bf(c[v]);
    }
}

// ---------------------------------------------------------------------------
// Launch: inputs are (x, Wq, Wk, Wv, Wo, timelike_mask); d_out = out ++ probs.
// ---------------------------------------------------------------------------
extern "C" void kernel_launch(void* const* d_in, const int* in_sizes, int n_in,
                              void* d_out, int out_size, void* d_ws, size_t ws_size,
                              hipStream_t stream)
{
    (void)in_sizes; (void)n_in; (void)out_size; (void)ws_size;
    const int B = 2, L = 2048, D = 1024, H = 16, dh = 64;
    const int M = B * L;                               // 4096 token rows

    const float* x    = (const float*)d_in[0];
    const float* Wq   = (const float*)d_in[1];
    const float* Wk   = (const float*)d_in[2];
    const float* Wv   = (const float*)d_in[3];
    const float* Wo   = (const float*)d_in[4];
    const int*   mask = (const int*)d_in[5];

    const size_t tile = (size_t)M * D;                 // elements per bf16 buffer
    __bf16* Qb   = (__bf16*)d_ws;
    __bf16* Kb   = Qb   + tile;
    __bf16* Vb   = Kb   + tile;
    __bf16* Qmod = Vb   + tile;
    __bf16* Vt   = Qmod + tile;
    __bf16* Ob   = Vt   + tile;                        // 48 MB workspace total

    float* out   = (float*)d_out;
    float* probs = out + (size_t)M * D;

    const dim3 gblk(128);
    const dim3 ggrd(D / 64, M / 16);
    gemm_xwT_wmma<float, __bf16><<<ggrd, gblk, 0, stream>>>(x, Wq, Qb, M, D, D);
    gemm_xwT_wmma<float, __bf16><<<ggrd, gblk, 0, stream>>>(x, Wk, Kb, M, D, D);
    gemm_xwT_wmma<float, __bf16><<<ggrd, gblk, 0, stream>>>(x, Wv, Vb, M, D, D);

    qmod_kernel<<<(M * H * 32) / 256, 256, 0, stream>>>(Qb, mask, Qmod, M * H, H, dh, D);

    v_transpose<<<(B * H * dh * L) / 256, 256, 0, stream>>>(Vb, Vt, B, H, L, dh, D);

    const size_t attn_lds = (size_t)16 * L * sizeof(float) + 8 * 4096;  // 160 KB
    lorentz_attn<<<B * H * (L / 16), 256, attn_lds, stream>>>(
        Qmod, Kb, Vt, probs, Ob, B, H, L, dh, D);

    gemm_xwT_wmma<__bf16, float><<<ggrd, gblk, 0, stream>>>(Ob, Wo, out, M, D, D);
}
